// Superpixel_SAT_24223615550148
// MI455X (gfx1250) — matrix operand order repaired and checked by reading, not compiled, not run on hardware
//
#include <hip/hip_runtime.h>
#include <stdint.h>

typedef __attribute__((ext_vector_type(16))) _Float16 v16h;
typedef __attribute__((ext_vector_type(8)))  float    v8f;

#define HSTRIDE 32   // padded row stride (floats) of per-head h buffers

// ---------------- float <-> order-preserving uint (for atomic max on f32) ---
__device__ __forceinline__ unsigned f2ord(float x) {
  unsigned u = __float_as_uint(x);
  return (u & 0x80000000u) ? ~u : (u | 0x80000000u);
}
__device__ __forceinline__ float ord2f(unsigned k) {
  unsigned u = (k & 0x80000000u) ? (k & 0x7fffffffu) : ~k;
  return __uint_as_float(u);
}

// ---------------- prepack W (+zero padding) into B-fragment order, f16 ------
// Wp layout: [kchunk][ct][lane][16 halfs]  -> each lane's fragment is one
// contiguous 32B run, so the GEMM kernel loads B with plain b128s.
template <int DIN, int DOUT>
__global__ __launch_bounds__(64) void pack_w_kernel(
    const float* __restrict__ W, _Float16* __restrict__ Wp)
{
  const int t = threadIdx.x;           // 64 threads: ct = t>>5, lane = t&31
  const int ct = t >> 5, lane = t & 31;
  const int half = lane >> 4, mr = lane & 15;
  const int col = ct * 16 + mr;
  constexpr int KCH = (DIN + 31) / 32;
  #pragma unroll
  for (int kc = 0; kc < KCH; ++kc) {
    #pragma unroll
    for (int j = 0; j < 16; ++j) {
      const int k = kc * 32 + half * 8 + j + (j >= 8 ? 8 : 0);
      const float v = (col < DOUT && k < DIN) ? W[k * DOUT + col] : 0.0f;
      Wp[((kc * 2 + ct) * 32 + lane) * 16 + j] = (_Float16)v;
    }
  }
}

// ---------------- dense: h = X@W + b ; a1 = h.a1v + a1b ; a2 = h.a2v + a2b --
// Grid-stride over 16-row tiles, one wave per tile iteration.
// f16 WMMA (16x16x32), f32 accumulation. B fragments hoisted out of the loop.
template <int DIN, int DOUT>
__global__ __launch_bounds__(256) void sat_linear_kernel(
    const float* __restrict__ X, int n, int ntiles,
    const _Float16* __restrict__ Wp, const float* __restrict__ b,
    const float* __restrict__ a1v, const float* __restrict__ a1b,
    const float* __restrict__ a2v, const float* __restrict__ a2b,
    float* __restrict__ h_out,          // [n, HSTRIDE] padded
    float* __restrict__ a1_out, float* __restrict__ a2_out)
{
  const int lane = threadIdx.x & 31;
  const int half = lane >> 4;                // 0: lanes 0-15, 1: lanes 16-31
  const int mr   = lane & 15;

  // ---- loop-invariant B fragments (32B contiguous each) + column params ----
  v16h bfr0[2], bfr1[2];
  float bias[2], w1[2], w2[2];
  #pragma unroll
  for (int ct = 0; ct < 2; ++ct) {
    bfr0[ct] = *(const v16h*)(Wp + ((0 * 2 + ct) * 32 + lane) * 16);
    if (DIN > 32) bfr1[ct] = *(const v16h*)(Wp + ((1 * 2 + ct) * 32 + lane) * 16);
    const int  col  = ct * 16 + mr;
    const int  colc = (col < DOUT) ? col : (DOUT - 1);
    const bool cv   = (col < DOUT);
    bias[ct] = cv ? b[colc]   : 0.0f;
    w1[ct]   = cv ? a1v[colc] : 0.0f;
    w2[ct]   = cv ? a2v[colc] : 0.0f;
  }
  const float ab1 = a1b[0], ab2 = a2b[0];

  const int wstride = gridDim.x * (blockDim.x >> 5);
  for (int tile = blockIdx.x * (blockDim.x >> 5) + (threadIdx.x >> 5);
       tile < ntiles; tile += wstride) {
    const int  r0   = tile * 16;
    const bool full = (r0 + 16 <= n);        // wave-uniform fast path

    // ---- A fragment: lane(l%16)=row, elem j -> K = 8*half + j + (j>=8?8:0)
    // Contiguous runs pa[0..7], pa[16..23]; all float4s 16B aligned.
    int arow = r0 + mr; if (arow >= n) arow = n - 1;
    const float* xrow = X + (size_t)arow * DIN;
    const float* pa = xrow + half * 8;
    const float4 x0 = *(const float4*)(pa + 0);
    const float4 x1 = *(const float4*)(pa + 4);
    const float4 x2 = *(const float4*)(pa + 16);
    const float4 x3 = *(const float4*)(pa + 20);
    v16h afr0;
    afr0[0]=(_Float16)x0.x; afr0[1]=(_Float16)x0.y; afr0[2]=(_Float16)x0.z; afr0[3]=(_Float16)x0.w;
    afr0[4]=(_Float16)x1.x; afr0[5]=(_Float16)x1.y; afr0[6]=(_Float16)x1.z; afr0[7]=(_Float16)x1.w;
    afr0[8]=(_Float16)x2.x; afr0[9]=(_Float16)x2.y; afr0[10]=(_Float16)x2.z; afr0[11]=(_Float16)x2.w;
    afr0[12]=(_Float16)x3.x; afr0[13]=(_Float16)x3.y; afr0[14]=(_Float16)x3.z; afr0[15]=(_Float16)x3.w;

    v16h afr1 = {};                          // K = 32..39 chunk (DIN==40 only)
    if (DIN > 32) {
      const float* pb = xrow + (half == 0 ? 32 : 0);   // half==1 addr is dummy
      const float4 y0 = *(const float4*)(pb + 0);
      const float4 y1 = *(const float4*)(pb + 4);
      const float z = 0.0f;
      afr1[0]=(_Float16)(half==0 ? y0.x : z); afr1[1]=(_Float16)(half==0 ? y0.y : z);
      afr1[2]=(_Float16)(half==0 ? y0.z : z); afr1[3]=(_Float16)(half==0 ? y0.w : z);
      afr1[4]=(_Float16)(half==0 ? y1.x : z); afr1[5]=(_Float16)(half==0 ? y1.y : z);
      afr1[6]=(_Float16)(half==0 ? y1.z : z); afr1[7]=(_Float16)(half==0 ? y1.w : z);
    }

    float p1[8], p2[8];
    #pragma unroll
    for (int i = 0; i < 8; ++i) { p1[i] = 0.0f; p2[i] = 0.0f; }

    #pragma unroll
    for (int ct = 0; ct < 2; ++ct) {
      v8f c = {};
      c = __builtin_amdgcn_wmma_f32_16x16x32_f16(false, afr0, false, bfr0[ct],
                                                 (short)0, c, false, false);
      if (DIN > 32)
        c = __builtin_amdgcn_wmma_f32_16x16x32_f16(false, afr1, false, bfr1[ct],
                                                   (short)0, c, false, false);
      // D layout: elem i -> row r0 + i + 8*half, col = ct*16 + mr
      if (full) {
        // one base pointer, stores at compile-time immediate offsets (i*128B)
        float* hp = h_out + (size_t)(r0 + 8 * half) * HSTRIDE + ct * 16 + mr;
        #pragma unroll
        for (int i = 0; i < 8; ++i) {
          const float val = c[i] + bias[ct];
          hp[i * HSTRIDE] = val;
          p1[i] += val * w1[ct];
          p2[i] += val * w2[ct];
        }
      } else {
        #pragma unroll
        for (int i = 0; i < 8; ++i) {
          int row = r0 + i + 8 * half;
          if (row >= n) row = n - 1;         // duplicates write same value
          const float val = c[i] + bias[ct];
          h_out[(size_t)row * HSTRIDE + ct * 16 + mr] = val;
          p1[i] += val * w1[ct];
          p2[i] += val * w2[ct];
        }
      }
    }
    // reduce across the 16 lanes of each half-wave (cols) -> a1/a2 per row
    #pragma unroll
    for (int i = 0; i < 8; ++i) {
      float s1 = p1[i], s2 = p2[i];
      #pragma unroll
      for (int d = 1; d < 16; d <<= 1) {
        s1 += __shfl_xor(s1, d, 16);
        s2 += __shfl_xor(s2, d, 16);
      }
      if (mr == 0) {
        int row = r0 + i + 8 * half;
        if (row >= n) row = n - 1;
        a1_out[row] = s1 + ab1;
        a2_out[row] = s2 + ab2;
      }
    }
  }
}

// ---------------- edge pass 1: segment max of leaky_relu(a1[r]+a2[c]) -------
__global__ __launch_bounds__(256) void edge_max_kernel(
    const int* __restrict__ rows, const int* __restrict__ cols,
    const float* __restrict__ a1, const float* __restrict__ a2,
    unsigned* __restrict__ m, int E)
{
  const int e = blockIdx.x * blockDim.x + threadIdx.x;
  if (e >= E) return;
  const int r = rows[e], c = cols[e];
  float v = a1[r] + a2[c];
  v = (v > 0.0f) ? v : 0.01f * v;
  atomicMax(m + r, f2ord(v));
}

// ---------------- edge pass 2: s[r] += e ; acc[r,:] += e * h[c,:] ----------
template <int DOUT>
__global__ __launch_bounds__(256) void edge_accum_kernel(
    const int* __restrict__ rows, const int* __restrict__ cols,
    const float* __restrict__ a1, const float* __restrict__ a2,
    const unsigned* __restrict__ m, const float* __restrict__ h,
    float* __restrict__ s, float* __restrict__ acc, int E)
{
  const int e = blockIdx.x * blockDim.x + threadIdx.x;
  if (e >= E) return;
  const int r = rows[e], c = cols[e];
  float v = a1[r] + a2[c];
  v = (v > 0.0f) ? v : 0.01f * v;
  const float ex = __expf(v - ord2f(m[r]));
  atomicAdd(s + r, ex);
  const float* hc = h + (size_t)c * HSTRIDE;   // 128B-strided rows, 16B aligned
  float* ar = acc + (size_t)r * DOUT;
  #pragma unroll
  for (int k = 0; k < DOUT; k += 4) {
    const float4 hv = *(const float4*)(hc + k);
    atomicAdd(ar + k + 0, ex * hv.x);
    atomicAdd(ar + k + 1, ex * hv.y);
    atomicAdd(ar + k + 2, ex * hv.z);
    atomicAdd(ar + k + 3, ex * hv.w);
  }
}

// ---------------- out[i, coff+k] = relu(acc[i,k]/s[i]) (layers 0,1) ---------
template <int DOUT>
__global__ __launch_bounds__(256) void normalize_kernel(
    const float* __restrict__ acc, const float* __restrict__ s,
    float* __restrict__ out, int n, int ostride, int coff)
{
  const int t = blockIdx.x * blockDim.x + threadIdx.x;
  if (t >= n * DOUT) return;
  const int i = t / DOUT, k = t - i * DOUT;
  const float v = fmaxf(acc[t] / s[i], 0.0f);
  out[(size_t)i * ostride + coff + k] = v;
}

// ---------------- layer-2 combine: (acc0/s0 + acc1/s1) / 2 ------------------
__global__ __launch_bounds__(256) void combine3_kernel(
    const float* __restrict__ acc0, const float* __restrict__ s0,
    const float* __restrict__ acc1, const float* __restrict__ s1,
    float* __restrict__ out, int n)
{
  const int t = blockIdx.x * blockDim.x + threadIdx.x;
  if (t >= n * 32) return;
  const int i = t >> 5;
  out[t] = 0.5f * (acc0[t] / s0[i] + acc1[t] / s1[i]);
}

// ---------------- global mean pool (accumulate) -----------------------------
__global__ __launch_bounds__(256) void pool_kernel(
    const float* __restrict__ x, const int* __restrict__ batch,
    float* __restrict__ pool, float* __restrict__ cnt, int n)
{
  const int t = blockIdx.x * blockDim.x + threadIdx.x;
  if (t >= n * 32) return;
  const int i = t >> 5, k = t & 31;
  const int g = batch[i];
  atomicAdd(pool + g * 32 + k, x[t]);
  if (k == 0) atomicAdd(cnt + g, 1.0f);
}

// ---------------- final linear (96->32) + row softmax -----------------------
__global__ __launch_bounds__(32) void final_kernel(
    const float* __restrict__ pool, const float* __restrict__ cnt,
    const float* __restrict__ Wf, const float* __restrict__ bf,
    float* __restrict__ out)
{
  const int g = blockIdx.x;     // 64 graphs
  const int j = threadIdx.x;    // 32 out columns
  float acc = bf[j];
  for (int br = 0; br < 3; ++br) {
    const float inv = 1.0f / fmaxf(cnt[br * 64 + g], 1.0f);
    for (int f = 0; f < 32; ++f) {
      const float x = pool[(br * 64 + g) * 32 + f] * inv;
      acc += x * Wf[(br * 32 + f) * 32 + j];
    }
  }
  float mx = acc;
  #pragma unroll
  for (int d = 1; d < 32; d <<= 1) mx = fmaxf(mx, __shfl_xor(mx, d, 32));
  const float ex = __expf(acc - mx);
  float sum = ex;
  #pragma unroll
  for (int d = 1; d < 32; d <<= 1) sum += __shfl_xor(sum, d, 32);
  out[g * 32 + j] = ex / sum;
}

// ============================================================================
extern "C" void kernel_launch(void* const* d_in, const int* in_sizes, int n_in,
                              void* d_out, int out_size, void* d_ws, size_t ws_size,
                              hipStream_t stream) {
  if (n_in < 120) return;

  const float* Xs[3] = {(const float*)d_in[0], (const float*)d_in[1], (const float*)d_in[2]};
  const int n[3] = {in_sizes[0] / 32, in_sizes[1] / 32, in_sizes[2] / 32};

  // Top-level ordering: dict-insertion (edges then batches) vs sorted pytree
  // (batches then edges; edge1d before edge1u). Detect via in_sizes[3].
  int iE0, iE1u, iE1d, iE2, iB0, iB1, iB2;
  if (in_sizes[3] == n[0]) { iB0 = 3; iB1 = 4; iB2 = 5; iE0 = 6; iE1d = 7; iE1u = 8; iE2 = 9; }
  else                     { iE0 = 3; iE1u = 4; iE1d = 5; iE2 = 6; iB0 = 7; iB1 = 8; iB2 = 9; }
  const int* batch[3] = {(const int*)d_in[iB0], (const int*)d_in[iB1], (const int*)d_in[iB2]};
  const int  eidx[3][2] = {{iE0, iE0}, {iE1u, iE1d}, {iE2, iE2}};

  // Param leaves start at index 10. SAT dict order: insertion (W,b,a1,a1b,a2,a2b)
  // vs sorted pytree (W,a1,a1b,a2,a2b,b). Detect via third leaf size.
  struct SatP { const float *W, *b, *a1, *a1b, *a2, *a2b; };
  const bool sortedLeaf = (in_sizes[12] == 1);
  SatP P[3][3][2];
  int idx = 10;
  for (int br = 0; br < 3; ++br)
    for (int l = 0; l < 3; ++l)
      for (int hh = 0; hh < 2; ++hh) {
        void* const* p = d_in + idx;
        SatP& sp = P[br][l][hh];
        sp.W = (const float*)p[0];
        if (sortedLeaf) {
          sp.a1 = (const float*)p[1]; sp.a1b = (const float*)p[2];
          sp.a2 = (const float*)p[3]; sp.a2b = (const float*)p[4];
          sp.b  = (const float*)p[5];
        } else {
          sp.b  = (const float*)p[1];
          sp.a1 = (const float*)p[2]; sp.a1b = (const float*)p[3];
          sp.a2 = (const float*)p[4]; sp.a2b = (const float*)p[5];
        }
        idx += 6;
      }
  const float* Wf = (const float*)d_in[idx];
  const float* bf = (const float*)d_in[idx + 1];

  // ---- workspace carve (all offsets 256B aligned) --------------------------
  size_t nm = n[0]; if ((size_t)n[1] > nm) nm = n[1]; if ((size_t)n[2] > nm) nm = n[2];
  char* w = (char*)d_ws;
  auto carve = [&](size_t bytes) { char* p = w; w += (bytes + 255) & ~(size_t)255; return p; };
  float*     buf1 = (float*)carve(nm * 40 * 4);  // layer0 out / layer2 out (out3)
  float*     buf2 = (float*)carve(nm * 40 * 4);  // layer1 out
  float*     hbuf[2]; hbuf[0] = (float*)carve(nm * HSTRIDE * 4); hbuf[1] = (float*)carve(nm * HSTRIDE * 4);
  float*     abuf[2]; abuf[0] = (float*)carve(nm * 32 * 4); abuf[1] = (float*)carve(nm * 32 * 4);
  float*     a1b_[2]; a1b_[0] = (float*)carve(nm * 4); a1b_[1] = (float*)carve(nm * 4);
  float*     a2b_[2]; a2b_[0] = (float*)carve(nm * 4); a2b_[1] = (float*)carve(nm * 4);
  unsigned*  mb_[2];  mb_[0]  = (unsigned*)carve(nm * 4); mb_[1] = (unsigned*)carve(nm * 4);
  float*     sb_[2];  sb_[0]  = (float*)carve(nm * 4); sb_[1] = (float*)carve(nm * 4);
  _Float16*  wpack[2]; wpack[0] = (_Float16*)carve(4096); wpack[1] = (_Float16*)carve(4096);
  float*     pool = (float*)carve(3 * 64 * 32 * 4);
  float*     cnt  = (float*)carve(3 * 64 * 4);

  hipMemsetAsync(pool, 0, 3 * 64 * 32 * 4, stream);
  hipMemsetAsync(cnt,  0, 3 * 64 * 4, stream);

  for (int br = 0; br < 3; ++br) {
    const int nn = n[br];
    const int tiles  = (nn + 15) / 16;
    int blocks = (tiles + 7) / 8;              // 8 waves / 256-thread block
    if (blocks > 1536) blocks = 1536;          // grid-stride: reuse B in regs
    for (int l = 0; l < 3; ++l) {
      const float* in = (l == 0) ? Xs[br] : ((l == 1) ? buf1 : buf2);
      const int dout = (l == 2) ? 32 : 20;
      for (int hh = 0; hh < 2; ++hh) {
        const SatP& sp = P[br][l][hh];
        const int ie = eidx[br][hh];
        const int E  = in_sizes[ie] / 2;
        const int* ep   = (const int*)d_in[ie];
        const int* rows = ep;
        const int* cols = ep + E;

        if (l == 0) {
          pack_w_kernel<32, 20><<<1, 64, 0, stream>>>(sp.W, wpack[hh]);
          sat_linear_kernel<32, 20><<<blocks, 256, 0, stream>>>(
              in, nn, tiles, wpack[hh], sp.b, sp.a1, sp.a1b, sp.a2, sp.a2b,
              hbuf[hh], a1b_[hh], a2b_[hh]);
        } else if (l == 1) {
          pack_w_kernel<40, 20><<<1, 64, 0, stream>>>(sp.W, wpack[hh]);
          sat_linear_kernel<40, 20><<<blocks, 256, 0, stream>>>(
              in, nn, tiles, wpack[hh], sp.b, sp.a1, sp.a1b, sp.a2, sp.a2b,
              hbuf[hh], a1b_[hh], a2b_[hh]);
        } else {
          pack_w_kernel<40, 32><<<1, 64, 0, stream>>>(sp.W, wpack[hh]);
          sat_linear_kernel<40, 32><<<blocks, 256, 0, stream>>>(
              in, nn, tiles, wpack[hh], sp.b, sp.a1, sp.a1b, sp.a2, sp.a2b,
              hbuf[hh], a1b_[hh], a2b_[hh]);
        }

        hipMemsetAsync(mb_[hh], 0, (size_t)nn * 4, stream);   // 0 == ordered(-inf)
        hipMemsetAsync(sb_[hh], 0, (size_t)nn * 4, stream);
        hipMemsetAsync(abuf[hh], 0, (size_t)nn * dout * 4, stream);

        edge_max_kernel<<<(E + 255) / 256, 256, 0, stream>>>(
            rows, cols, a1b_[hh], a2b_[hh], mb_[hh], E);
        if (l < 2)
          edge_accum_kernel<20><<<(E + 255) / 256, 256, 0, stream>>>(
              rows, cols, a1b_[hh], a2b_[hh], mb_[hh], hbuf[hh],
              sb_[hh], abuf[hh], E);
        else
          edge_accum_kernel<32><<<(E + 255) / 256, 256, 0, stream>>>(
              rows, cols, a1b_[hh], a2b_[hh], mb_[hh], hbuf[hh],
              sb_[hh], abuf[hh], E);

        if (l < 2) {
          float* outb = (l == 0) ? buf1 : buf2;
          const int tot = nn * 20;
          normalize_kernel<20><<<(tot + 255) / 256, 256, 0, stream>>>(
              abuf[hh], sb_[hh], outb, nn, 40, hh * 20);
        }
      }
      if (l == 2) {
        combine3_kernel<<<(nn * 32 + 255) / 256, 256, 0, stream>>>(
            abuf[0], sb_[0], abuf[1], sb_[1], buf1 /* out3 reuses buf1 */, nn);
      }
    }
    pool_kernel<<<(nn * 32 + 255) / 256, 256, 0, stream>>>(
        buf1, batch[br], pool + br * 64 * 32, cnt + br * 64, nn);
  }

  final_kernel<<<64, 32, 0, stream>>>(pool, cnt, Wf, bf, (float*)d_out);
  (void)out_size; (void)ws_size;
}
